// GIN_17128329576798
// MI455X (gfx1250) — matrix operand order repaired and checked by reading, not compiled, not run on hardware
//
#include <hip/hip_runtime.h>
#include <stdint.h>

// ---------------- problem sizes (fixed by reference) ----------------
#define BS    4096
#define NODES 32
#define NTOT  (BS * NODES)     // 131072
#define F_IN  128
#define EPG   512
#define ETOT  (BS * EPG)       // 2097152
#define HID   1024
#define NCLS  10
#define KDIM  (F_IN * NODES)   // 4096  (lin1 K)

// ---------------- WMMA types ----------------
typedef __attribute__((ext_vector_type(16))) __bf16    v16bf;
typedef __attribute__((ext_vector_type(8)))  float     v8f;
typedef __attribute__((ext_vector_type(4)))  uint32_t  u32x4;
typedef __attribute__((ext_vector_type(8)))  uint32_t  u32x8;

union Frag {
    uint32_t       u32[8];
    unsigned short u16[16];
    uint4          q[2];
    v16bf          v;
};

static __device__ __forceinline__ unsigned short f2bf(float f) {
    // round-to-nearest-even fp32 -> bf16
    uint32_t u = __float_as_uint(f);
    u += 0x7FFFu + ((u >> 16) & 1u);
    return (unsigned short)(u >> 16);
}
static __device__ __forceinline__ float bf2f(unsigned short h) {
    return __uint_as_float(((uint32_t)h) << 16);
}
static __device__ __forceinline__ v8f v8f_zero() {
    v8f z = {0.f, 0.f, 0.f, 0.f, 0.f, 0.f, 0.f, 0.f};
    return z;
}
static __device__ __forceinline__ uint32_t lds_off_of(const void* p) {
    // LDS flat addresses keep the LDS byte offset in addr[31:0] (aperture in high bits)
    return (uint32_t)(uintptr_t)p;
}

// ---- per-lane async global->LDS copy (ASYNCcnt path), 16 bytes per lane ----
static __device__ __forceinline__ void async_g2l_b128(uint32_t lds_off, const void* gptr) {
    asm volatile("global_load_async_to_lds_b128 %0, %1, off"
                 :: "v"(lds_off), "v"((uint64_t)(uintptr_t)gptr)
                 : "memory");
}
static __device__ __forceinline__ void wait_async0() {
    asm volatile("s_wait_asynccnt 0" ::: "memory");
}

// ---- Tensor Data Mover: 2D tile (rows x k) of bf16, rows contiguous in LDS ----
// D# per CDNA5 ISA. data_size=1 (2 bytes/elem).
static __device__ __forceinline__ void tdm_load_2d_bf16(
    uint32_t lds_byte_addr, const void* gaddr,
    uint32_t tile_k,          // tile_dim0 (elements)
    uint32_t tile_rows,       // tile_dim1
    uint32_t row_stride,      // tensor_dim0_stride (elements)
    uint32_t tensor_k,        // tensor_dim0 (elements)
    uint32_t tensor_rows)     // tensor_dim1
{
    const uint64_t ga = (uint64_t)(uintptr_t)gaddr;
    u32x4 g0;
    g0[0] = 1u;                                   // count=1, user descriptor
    g0[1] = lds_byte_addr;                        // lds_addr
    g0[2] = (uint32_t)ga;                         // global_addr[31:0]
    g0[3] = (uint32_t)(ga >> 32) | (2u << 30);    // global_addr[56:32] | type=2
    u32x8 g1;
    g1[0] = (1u << 16);                           // workgroup_mask=0, data_size=1 (2B)
    g1[1] = (tensor_k & 0xFFFFu) << 16;           // tensor_dim0[15:0]
    g1[2] = (tensor_k >> 16) | ((tensor_rows & 0xFFFFu) << 16);
    g1[3] = (tensor_rows >> 16) | ((tile_k & 0xFFFFu) << 16);  // tile_dim0
    g1[4] = (tile_rows & 0xFFFFu);                // tile_dim1 (tile_dim2=0)
    g1[5] = row_stride;                           // tensor_dim0_stride[31:0]
    g1[6] = 0;                                    // stride[47:32], dim1_stride[15:0]
    g1[7] = 0;
    u32x4 g2 = {0, 0, 0, 0};                      // tensor_dim2/3 unused
    u32x4 g3 = {0, 0, 0, 0};                      // tensor_dim3_stride/dim4 unused
    asm volatile("tensor_load_to_lds %0, %1, %2, %3"
                 :: "s"(g0), "s"(g1), "s"(g2), "s"(g3)
                 : "memory");
}

// ---------------- fp32 -> bf16 weight conversion ----------------
__global__ void cvt_f32_bf16(const float* __restrict__ src,
                             unsigned short* __restrict__ dst, int n) {
    int i = blockIdx.x * blockDim.x + threadIdx.x;
    if (i < n) dst[i] = f2bf(src[i]);
}

// ---------------- stage 1: GIN aggregation + conv GEMM, fused ----------------
// One block per graph (32 nodes). x tile async-loaded into LDS, edges
// accumulated with ds_add_f32, then 32x128x128 GEMM via v_wmma bf16, ReLU,
// stored directly in rearranged layout H[g][f*32 + node] as bf16.
__global__ __launch_bounds__(256) void gin_conv_kernel(
    const float* __restrict__ x,
    const long long* __restrict__ eidx,      // [2, ETOT] int64
    const unsigned short* __restrict__ Wg,   // [128][128] bf16 (out, in)
    const float* __restrict__ gb,            // [128]
    unsigned short* __restrict__ Hbuf)       // [BS][4096] bf16, col = f*32+node
{
    __shared__ float          agg[NODES][F_IN];   // 16 KB
    __shared__ unsigned short abf[NODES][F_IN];   // 8 KB

    const int g = blockIdx.x;
    const int t = threadIdx.x;
    const float* xg = x + (size_t)g * NODES * F_IN;

    // self term (eps = 0): agg = x  -- async copy global -> LDS (bypasses VGPRs)
    const uint32_t agg_base = lds_off_of(&agg[0][0]);
    #pragma unroll
    for (int i = 0; i < 4; ++i) {
        const int idx = t + 256 * i;   // 1024 x 16B = 16 KB
        async_g2l_b128(agg_base + (uint32_t)idx * 16,
                       (const char*)xg + (size_t)idx * 16);
    }
    wait_async0();
    __syncthreads();

    // edges of this graph are contiguous: [g*EPG, (g+1)*EPG)
    const long long* srcp = eidx + (size_t)g * EPG;
    const long long* dstp = eidx + (size_t)ETOT + (size_t)g * EPG;
    for (int e = t; e < EPG; e += 256) {
        const int sl = (int)srcp[e] & (NODES - 1);
        const int dl = (int)dstp[e] & (NODES - 1);
        const float4* xs = (const float4*)(xg + sl * F_IN);
        float* ad = &agg[dl][0];
        #pragma unroll
        for (int f4 = 0; f4 < F_IN / 4; ++f4) {
            float4 v = xs[f4];
            atomicAdd(&ad[f4 * 4 + 0], v.x);   // ds_add_f32
            atomicAdd(&ad[f4 * 4 + 1], v.y);
            atomicAdd(&ad[f4 * 4 + 2], v.z);
            atomicAdd(&ad[f4 * 4 + 3], v.w);
        }
    }
    __syncthreads();

    // convert accumulated tile to bf16
    for (int i = t; i < NODES * F_IN; i += 256)
        (&abf[0][0])[i] = f2bf((&agg[0][0])[i]);
    __syncthreads();

    // WMMA: out[m, n] = sum_k agg[m,k] * Wg[n,k];  M=32, N=128, K=128
    const int wave = t >> 5;           // 0..7  -> N tile
    const int lane = t & 31;
    const int half = lane >> 4;        // 0|1
    const int ln   = lane & 15;

    v8f c0 = v8f_zero();               // m rows 0..15
    v8f c1 = v8f_zero();               // m rows 16..31
    #pragma unroll
    for (int kk = 0; kk < F_IN; kk += 32) {
        Frag b;  // B 32x16: lane=col n; chunks at K = half*8 and 16+half*8
        const unsigned short* wr = Wg + (size_t)(wave * 16 + ln) * F_IN + kk + half * 8;
        b.q[0] = *(const uint4*)(wr);
        b.q[1] = *(const uint4*)(wr + 16);
        Frag a0, a1; // A 16x32: lane=row m; chunks at K = half*8 and 16+half*8
        a0.q[0] = *(const uint4*)&abf[ln][kk + half * 8];
        a0.q[1] = *(const uint4*)&abf[ln][kk + 16 + half * 8];
        a1.q[0] = *(const uint4*)&abf[16 + ln][kk + half * 8];
        a1.q[1] = *(const uint4*)&abf[16 + ln][kk + 16 + half * 8];
        c0 = __builtin_amdgcn_wmma_f32_16x16x32_bf16(false, a0.v, false, b.v,
                                                     (short)0, c0, false, false);
        c1 = __builtin_amdgcn_wmma_f32_16x16x32_bf16(false, a1.v, false, b.v,
                                                     (short)0, c1, false, false);
    }

    // epilogue: +bias, ReLU, bf16, store rearranged: H[g][n*32 + m]
    const int n = wave * 16 + ln;     // output feature index (fixed per lane)
    const float bias = gb[n];
    unsigned short* hrow = Hbuf + (size_t)g * KDIM;
    unsigned short tmp[8];
    #pragma unroll
    for (int r = 0; r < 8; ++r) tmp[r] = f2bf(fmaxf(c0[r] + bias, 0.f));
    *(uint4*)&hrow[n * NODES + 8 * half] = *(uint4*)tmp;          // m = 8*half + r
    #pragma unroll
    for (int r = 0; r < 8; ++r) tmp[r] = f2bf(fmaxf(c1[r] + bias, 0.f));
    *(uint4*)&hrow[n * NODES + 16 + 8 * half] = *(uint4*)tmp;     // m = 16 + 8*half + r
}

// -------- stage 2: lin1 GEMM 4096x4096x1024, TDM-staged + double-buffered --------
#define TB_M 128
#define TB_N 128
#define TB_K 64

__global__ __launch_bounds__(256) void lin1_kernel(
    const unsigned short* __restrict__ H,    // [4096][4096] bf16
    const unsigned short* __restrict__ W1,   // [1024][4096] bf16
    const float* __restrict__ b1,            // [1024]
    unsigned short* __restrict__ H1)         // [4096][1024] bf16
{
    __shared__ unsigned short sA[2][TB_M][TB_K];   // 32 KB
    __shared__ unsigned short sB[2][TB_N][TB_K];   // 32 KB

    const int t    = threadIdx.x;
    const int wave = t >> 5, lane = t & 31, half = lane >> 4, ln = lane & 15;
    const int m0 = (int)(blockIdx.x >> 3) * TB_M;   // 32 M-blocks
    const int n0 = (int)(blockIdx.x & 7)  * TB_N;   // 8  N-blocks
    const int wm = wave >> 2;   // 0..1 -> 64 rows
    const int wn = wave & 3;    // 0..3 -> 32 cols

    v8f acc[4][2];
    #pragma unroll
    for (int ms = 0; ms < 4; ++ms)
        #pragma unroll
        for (int ns = 0; ns < 2; ++ns) acc[ms][ns] = v8f_zero();

    const uint32_t sA0 = lds_off_of(&sA[0][0][0]);
    const uint32_t sA1 = lds_off_of(&sA[1][0][0]);
    const uint32_t sB0 = lds_off_of(&sB[0][0][0]);
    const uint32_t sB1 = lds_off_of(&sB[1][0][0]);

    const int nit = KDIM / TB_K;   // 64

    // prologue: stage tile 0 via TDM (wave 0 only; tracked on TENSORcnt)
    if (wave == 0) {
        tdm_load_2d_bf16(sA0, H  + (size_t)m0 * KDIM, TB_K, TB_M, KDIM, KDIM, BS);
        tdm_load_2d_bf16(sB0, W1 + (size_t)n0 * KDIM, TB_K, TB_N, KDIM, KDIM, HID);
    }

    for (int it = 0; it < nit; ++it) {
        const int cur = it & 1;
        if (wave == 0) {
            if (it + 1 < nit) {
                const int k1 = (it + 1) * TB_K;
                tdm_load_2d_bf16(cur ? sA0 : sA1, H  + (size_t)m0 * KDIM + k1,
                                 TB_K, TB_M, KDIM, KDIM, BS);
                tdm_load_2d_bf16(cur ? sB0 : sB1, W1 + (size_t)n0 * KDIM + k1,
                                 TB_K, TB_N, KDIM, KDIM, HID);
                // TDM completes in-order per wave: <=2 outstanding => tile `it` landed
                __builtin_amdgcn_s_wait_tensorcnt(2);
            } else {
                __builtin_amdgcn_s_wait_tensorcnt(0);
            }
        }
        // GL2 prefetch of the tile after next
        if (it + 2 < nit) {
            const int row = t & 127;
            const int k2 = (it + 2) * TB_K;
            __builtin_prefetch(H  + (size_t)(m0 + row) * KDIM + k2, 0, 1);
            __builtin_prefetch(W1 + (size_t)(n0 + row) * KDIM + k2, 0, 1);
        }
        __syncthreads();   // tile `it` visible to all waves

        #pragma unroll
        for (int kk = 0; kk < TB_K; kk += 32) {
            Frag a[4], b[2];
            #pragma unroll
            for (int ms = 0; ms < 4; ++ms) {
                const int m = wm * 64 + ms * 16 + ln;
                a[ms].q[0] = *(const uint4*)&sA[cur][m][kk + half * 8];
                a[ms].q[1] = *(const uint4*)&sA[cur][m][kk + 16 + half * 8];
            }
            #pragma unroll
            for (int ns = 0; ns < 2; ++ns) {
                const int n = wn * 32 + ns * 16 + ln;
                b[ns].q[0] = *(const uint4*)&sB[cur][n][kk + half * 8];
                b[ns].q[1] = *(const uint4*)&sB[cur][n][kk + 16 + half * 8];
            }
            #pragma unroll
            for (int ms = 0; ms < 4; ++ms)
                #pragma unroll
                for (int ns = 0; ns < 2; ++ns)
                    acc[ms][ns] = __builtin_amdgcn_wmma_f32_16x16x32_bf16(
                        false, a[ms].v, false, b[ns].v, (short)0, acc[ms][ns],
                        false, false);
        }
        __syncthreads();   // all reads of buf `cur` done before it is restaged
    }

    // epilogue: +bias, ReLU, bf16 store
    #pragma unroll
    for (int ns = 0; ns < 2; ++ns) {
        const int n = n0 + wn * 32 + ns * 16 + ln;
        const float bias = b1[n];
        #pragma unroll
        for (int ms = 0; ms < 4; ++ms) {
            #pragma unroll
            for (int r = 0; r < 8; ++r) {
                const int m = m0 + wm * 64 + ms * 16 + 8 * half + r;
                H1[(size_t)m * HID + n] = f2bf(fmaxf(acc[ms][ns][r] + bias, 0.f));
            }
        }
    }
}

// ---------------- stage 3: lin2 + softmax (one wave per graph) ----------------
__global__ __launch_bounds__(256) void lin2_softmax_kernel(
    const unsigned short* __restrict__ H1,   // [4096][1024] bf16
    const unsigned short* __restrict__ W2,   // [10][1024] bf16
    const float* __restrict__ b2,            // [10]
    float* __restrict__ out)                 // [4096][10]
{
    const int t = threadIdx.x, wave = t >> 5, lane = t & 31;
    const int g = blockIdx.x * 8 + wave;
    const unsigned short* hr = H1 + (size_t)g * HID;

    float acc[NCLS];
    #pragma unroll
    for (int c = 0; c < NCLS; ++c) acc[c] = 0.f;

    for (int j = lane; j < HID; j += 32) {
        const float hv = bf2f(hr[j]);
        #pragma unroll
        for (int c = 0; c < NCLS; ++c)
            acc[c] += hv * bf2f(W2[c * HID + j]);
    }
    #pragma unroll
    for (int c = 0; c < NCLS; ++c)
        #pragma unroll
        for (int off = 16; off > 0; off >>= 1)
            acc[c] += __shfl_xor(acc[c], off, 32);

    if (lane == 0) {
        float lg[NCLS], mx = -1e30f, s = 0.f;
        #pragma unroll
        for (int c = 0; c < NCLS; ++c) { lg[c] = acc[c] + b2[c]; mx = fmaxf(mx, lg[c]); }
        #pragma unroll
        for (int c = 0; c < NCLS; ++c) { lg[c] = __expf(lg[c] - mx); s += lg[c]; }
        const float inv = 1.f / s;
        #pragma unroll
        for (int c = 0; c < NCLS; ++c) out[(size_t)g * NCLS + c] = lg[c] * inv;
    }
}

// ---------------- launcher ----------------
extern "C" void kernel_launch(void* const* d_in, const int* in_sizes, int n_in,
                              void* d_out, int out_size, void* d_ws, size_t ws_size,
                              hipStream_t stream) {
    const float*     x    = (const float*)d_in[0];
    const long long* eidx = (const long long*)d_in[1];
    const float*     gw   = (const float*)d_in[2];
    const float*     gb   = (const float*)d_in[3];
    const float*     w1   = (const float*)d_in[4];
    const float*     b1   = (const float*)d_in[5];
    const float*     w2   = (const float*)d_in[6];
    const float*     b2   = (const float*)d_in[7];
    float*           out  = (float*)d_out;

    // workspace carve-out (all bf16): H (32MB) | H1 (8MB) | Wg | W1 (8MB) | W2
    unsigned short* Hbuf = (unsigned short*)d_ws;
    unsigned short* H1   = Hbuf + (size_t)BS * KDIM;
    unsigned short* Wgb  = H1   + (size_t)BS * HID;
    unsigned short* W1b  = Wgb  + (size_t)F_IN * F_IN;
    unsigned short* W2b  = W1b  + (size_t)HID * KDIM;

    cvt_f32_bf16<<<(F_IN * F_IN + 255) / 256, 256, 0, stream>>>(gw, Wgb, F_IN * F_IN);
    cvt_f32_bf16<<<(HID * KDIM + 255) / 256, 256, 0, stream>>>(w1, W1b, HID * KDIM);
    cvt_f32_bf16<<<(NCLS * HID + 255) / 256, 256, 0, stream>>>(w2, W2b, NCLS * HID);

    gin_conv_kernel<<<BS, 256, 0, stream>>>(x, eidx, Wgb, gb, Hbuf);
    lin1_kernel<<<dim3(32 * 8), 256, 0, stream>>>(Hbuf, W1b, b1, H1);
    lin2_softmax_kernel<<<BS / 8, 256, 0, stream>>>(H1, W2b, b2, out);
}
